// DotProductAttention_240518168758
// MI455X (gfx1250) — matrix operand order repaired and checked by reading, not compiled, not run on hardware
//
#include <hip/hip_runtime.h>
#include <hip/hip_bf16.h>

typedef __attribute__((ext_vector_type(16))) _Float16 v16h;
typedef __attribute__((ext_vector_type(8)))  _Float16 v8h;
typedef __attribute__((ext_vector_type(2)))  _Float16 h2;
typedef __attribute__((ext_vector_type(8)))  float    v8f;

union U16H { v16h v; _Float16 h[16]; };
union U8F  { v8f  v; float f[8]; };

#define B_  32
#define Q_  2048
#define K_  2048
#define D_  128

#define SHUF16(a, b) __builtin_shufflevector((a), (b), 0,1,2,3,4,5,6,7,8,9,10,11,12,13,14,15)

// DPP16 lane permute (stays within a 16-lane row): ctrl must be a literal.
#define DPPF(x, ctrl) \
    __int_as_float(__builtin_amdgcn_update_dpp(0, __float_as_int(x), (ctrl), 0xf, 0xf, true))

static __device__ __forceinline__ float rowmax16(float x) {
    x = fmaxf(x, DPPF(x, 0xb1));   // quad_perm [1,0,3,2]
    x = fmaxf(x, DPPF(x, 0x4e));   // quad_perm [2,3,0,1]
    x = fmaxf(x, DPPF(x, 0x141));  // row_half_mirror
    x = fmaxf(x, DPPF(x, 0x140));  // row_mirror
    return x;
}
static __device__ __forceinline__ float rowsum16(float x) {
    x += DPPF(x, 0xb1);
    x += DPPF(x, 0x4e);
    x += DPPF(x, 0x141);
    x += DPPF(x, 0x140);
    return x;
}

__global__ __launch_bounds__(256)
void DotProductAttention_240518168758_kernel(const float* __restrict__ q,
                                             const float* __restrict__ kmat,
                                             const float* __restrict__ vmat,
                                             const int*   __restrict__ valid_lens,
                                             float*       __restrict__ out) {
    // Double-buffered K/V staging tiles; single barrier per iteration.
    __shared__ __align__(16) _Float16 Klds[2][64][136];  // 34816 B
    __shared__ __align__(16) _Float16 Vt[2][128][72];    // 36864 B
    __shared__ __align__(16) _Float16 Plds[8][16][72];   // 18432 B

    const int b     = blockIdx.y;
    const int qbase = blockIdx.x * 128;
    const int tid   = threadIdx.x;
    const int wave  = tid >> 5;
    const int lane  = tid & 31;
    const int hi    = lane >> 4;
    const int ln    = lane & 15;

    const int vl  = valid_lens[b];
    const int nkt = (vl + 63) >> 6;

    // log2(e)/sqrt(128): softmax runs in the base-2 domain so v_exp_f32 needs
    // no preceding multiply.
    const float scale = 0.12751879526953198f;

    // ---- Q block (16 rows) as 4 A-fragments (16x32 f16), pre-scaled ----
    U16H qa[4];
    {
        const int qrow = qbase + wave * 16 + ln;
        const float* qrp = q + ((size_t)b * Q_ + qrow) * D_;
#pragma unroll
        for (int c = 0; c < 4; ++c) {
            const float* qp = qrp + 32 * c + 8 * hi;
#pragma unroll
            for (int j = 0; j < 8; ++j) {
                qa[c].h[j]     = (_Float16)(qp[j]      * scale);
                qa[c].h[8 + j] = (_Float16)(qp[16 + j] * scale);
            }
        }
    }

    float mr[8], lr[8];
    U8F o[8];
#pragma unroll
    for (int r = 0; r < 8; ++r) { mr[r] = -3.0e38f; lr[r] = 0.0f; }
#pragma unroll
    for (int dc = 0; dc < 8; ++dc)
#pragma unroll
        for (int r = 0; r < 8; ++r) o[dc].f[r] = 0.0f;

    // K staging: 4 threads per key row, 32 d columns each.
    const int tkr = tid >> 2;          // 0..63
    const int td0 = (tid & 3) * 32;    // 0,32,64,96
    // V staging: 2 adjacent key rows x 16 d columns -> packed transposed writes.
    const int vkr = (tid & 31) * 2;    // 0..62
    const int vd0 = (tid >> 5) * 16;   // 0..112

    const float* kpt = kmat + ((size_t)b * K_ + tkr) * D_ + td0;
    const float* vpt = vmat + ((size_t)b * K_ + vkr) * D_ + vd0;

    // Stage one 64-key tile (fp32 -> f16, packed 4B LDS stores) into buffer `buf`.
    auto stage = [&](int kb0, int buf) {
        const float4* kp4 = (const float4*)(kpt + (size_t)kb0 * D_);
        h2* krow = (h2*)&Klds[buf][tkr][td0];
#pragma unroll
        for (int g = 0; g < 8; ++g) {
            float4 kv = kp4[g];
            krow[2 * g + 0] = (h2){(_Float16)kv.x, (_Float16)kv.y};
            krow[2 * g + 1] = (h2){(_Float16)kv.z, (_Float16)kv.w};
        }
        const float* vp0 = vpt + (size_t)kb0 * D_;
        const float* vp1 = vp0 + D_;
#pragma unroll
        for (int g = 0; g < 4; ++g) {
            float4 x = ((const float4*)vp0)[g];
            float4 y = ((const float4*)vp1)[g];
            *(h2*)&Vt[buf][vd0 + 4 * g + 0][vkr] = (h2){(_Float16)x.x, (_Float16)y.x};
            *(h2*)&Vt[buf][vd0 + 4 * g + 1][vkr] = (h2){(_Float16)x.y, (_Float16)y.y};
            *(h2*)&Vt[buf][vd0 + 4 * g + 2][vkr] = (h2){(_Float16)x.z, (_Float16)y.z};
            *(h2*)&Vt[buf][vd0 + 4 * g + 3][vkr] = (h2){(_Float16)x.w, (_Float16)y.w};
        }
    };

    stage(0, 0);

    for (int kt = 0; kt < nkt; ++kt) {
        const int kb0 = kt * 64;
        const int cur = kt & 1;

        __syncthreads();  // staging of buf `cur` complete; prior reads of buf `cur^1` done

        if (kt + 1 < nkt) {
            stage(kb0 + 64, cur ^ 1);
            if (kt + 2 < nkt) {
                __builtin_prefetch(kpt + (size_t)(kb0 + 128) * D_, 0, 0);
                __builtin_prefetch(vpt + (size_t)(kb0 + 128) * D_, 0, 0);
            }
        }

        // ---- Scores: four 16x16 column tiles over 4 d-chunks ----
        v8f s[4];
#pragma unroll
        for (int j = 0; j < 4; ++j) s[j] = (v8f){};
#pragma unroll
        for (int c = 0; c < 4; ++c) {
            v16h bf[4];
#pragma unroll
            for (int j = 0; j < 4; ++j) {
                const v8h* p = (const v8h*)&Klds[cur][16 * j + ln][32 * c + 8 * hi];
                bf[j] = SHUF16(p[0], p[2]);
            }
#pragma unroll
            for (int j = 0; j < 4; ++j)
                s[j] = __builtin_amdgcn_wmma_f32_16x16x32_f16(false, qa[c].v, false, bf[j],
                                                              (short)0, s[j], false, false);
        }
        U8F us[4];
#pragma unroll
        for (int j = 0; j < 4; ++j) us[j].v = s[j];

        // ---- Online softmax over 64 columns (base-2 domain, DPP reductions) ----
#pragma unroll
        for (int r = 0; r < 8; ++r) {
            float a[4];
#pragma unroll
            for (int j = 0; j < 4; ++j)
                a[j] = (kb0 + 16 * j + ln < vl) ? us[j].f[r] : -3.0e38f;
            float t = fmaxf(fmaxf(a[0], a[1]), fmaxf(a[2], a[3]));
            t = rowmax16(t);
            float mnew  = fmaxf(mr[r], t);
            float alpha = __builtin_amdgcn_exp2f(mr[r] - mnew);
            float e0 = __builtin_amdgcn_exp2f(a[0] - mnew);
            float e1 = __builtin_amdgcn_exp2f(a[1] - mnew);
            float e2 = __builtin_amdgcn_exp2f(a[2] - mnew);
            float e3 = __builtin_amdgcn_exp2f(a[3] - mnew);
            float rs = rowsum16((e0 + e1) + (e2 + e3));
            lr[r] = lr[r] * alpha + rs;
            mr[r] = mnew;
#pragma unroll
            for (int dc = 0; dc < 8; ++dc) o[dc].f[r] *= alpha;
            const int prow = r + 8 * hi;
            Plds[wave][prow][ln]      = (_Float16)e0;
            Plds[wave][prow][16 + ln] = (_Float16)e1;
            Plds[wave][prow][32 + ln] = (_Float16)e2;
            Plds[wave][prow][48 + ln] = (_Float16)e3;
        }

        // ---- P (16x64) as two A-fragments (wave-private LDS round trip) ----
        v16h pf[2];
#pragma unroll
        for (int t = 0; t < 2; ++t) {
            const v8h* pl = (const v8h*)&Plds[wave][ln][32 * t + 8 * hi];
            const v8h* ph = (const v8h*)&Plds[wave][ln][32 * t + 16 + 8 * hi];
            pf[t] = SHUF16(pl[0], ph[0]);
        }

        // ---- O += P x V : 2 K-blocks x 8 d-chunks ----
#pragma unroll
        for (int dc = 0; dc < 8; ++dc) {
            const v8h* v0p = (const v8h*)&Vt[cur][dc * 16 + ln][8 * hi];
            v16h vf0 = SHUF16(v0p[0], v0p[2]);
            o[dc].v = __builtin_amdgcn_wmma_f32_16x16x32_f16(false, pf[0], false, vf0,
                                                             (short)0, o[dc].v, false, false);
            const v8h* v1p = (const v8h*)&Vt[cur][dc * 16 + ln][32 + 8 * hi];
            v16h vf1 = SHUF16(v1p[0], v1p[2]);
            o[dc].v = __builtin_amdgcn_wmma_f32_16x16x32_f16(false, pf[1], false, vf1,
                                                             (short)0, o[dc].v, false, false);
        }
    }

    // ---- Normalize and store O (fp32) ----
    {
        const int qrow = qbase + wave * 16 + 8 * hi;
        float inv[8];
#pragma unroll
        for (int r = 0; r < 8; ++r) inv[r] = 1.0f / lr[r];
#pragma unroll
        for (int dc = 0; dc < 8; ++dc)
#pragma unroll
            for (int r = 0; r < 8; ++r)
                out[((size_t)b * Q_ + qrow + r) * D_ + dc * 16 + ln] = o[dc].f[r] * inv[r];
    }
}

extern "C" void kernel_launch(void* const* d_in, const int* in_sizes, int n_in,
                              void* d_out, int out_size, void* d_ws, size_t ws_size,
                              hipStream_t stream) {
    (void)in_sizes; (void)n_in; (void)out_size; (void)d_ws; (void)ws_size;
    const float* q  = (const float*)d_in[0];
    const float* k  = (const float*)d_in[1];
    const float* v  = (const float*)d_in[2];
    const int*   vl = (const int*)d_in[3];
    float* out = (float*)d_out;

    dim3 grid(Q_ / 128, B_);
    dim3 block(256);
    DotProductAttention_240518168758_kernel<<<grid, block, 0, stream>>>(q, k, v, vl, out);
}
